// SceneGraphModel_72980084293697
// MI455X (gfx1250) — compile-verified, matrix operand order checked by reading, not run on hardware
//
#include <hip/hip_runtime.h>

typedef __attribute__((ext_vector_type(16))) _Float16 v16h;
typedef __attribute__((ext_vector_type(8)))  _Float16 v8h;
typedef __attribute__((ext_vector_type(8)))  float    v8f;

#define E_EDGES   2000000
#define N_TILES   (E_EDGES / 32)
#define NUM_REL   51
#define NWAVES    4
#define NTHREADS  (NWAVES * 32)

union V16U { v16h v; v8h h[2]; };

// Load a 16-half A/B fragment: two contiguous 8-half (16B) runs at row+k0 and row+k0+16.
// CDNA5 16-bit fragment layout (lane = m%16, khalf = lane/16): elements 0..7 hold
// K = k0..k0+7, elements 8..15 hold K = k0+16..k0+23 (k0 = kb*32 + 8*khalf).
__device__ __forceinline__ v16h load_frag(const _Float16* row, int k0) {
  V16U u;
  u.h[0] = *(const v8h*)(row + k0);
  u.h[1] = *(const v8h*)(row + k0 + 16);
  return u.v;
}

__device__ __forceinline__ v8f splat8(float x) {
  v8f c;
#pragma unroll
  for (int i = 0; i < 8; ++i) c[i] = x;
  return c;
}

// Wave-local LDS RAW fence + compiler barrier.
__device__ __forceinline__ void lds_fence() {
  asm volatile("s_wait_dscnt 0" ::: "memory");
}

__global__ __launch_bounds__(NTHREADS) void sg_mlp_kernel(
    const float* __restrict__ obj,     // [100000, 8]
    const int*   __restrict__ pairs,   // [E, 2]
    const float* __restrict__ oW0, const float* __restrict__ ob0,  // 16x128
    const float* __restrict__ oW1, const float* __restrict__ ob1,  // 128x64
    const float* __restrict__ oW2, const float* __restrict__ ob2,  // 64x32
    const float* __restrict__ oW3, const float* __restrict__ ob3,  // 32x16
    const float* __restrict__ rW0, const float* __restrict__ rb0,  // 16x8
    const float* __restrict__ rW1, const float* __restrict__ rb1,  // 8x4
    const float* __restrict__ rW2, const float* __restrict__ rb2,  // 4x51
    float* __restrict__ out)           // [E, 51]
{
  // ---- shared weights, transposed [N][K] in f16 (K padded to mult of 32 for L0;
  //      L0 row K=16 carries the bias, driven by a 1.0 in A) ----
  __shared__ __align__(16) _Float16 sWT0[128 * 32];   //  8 KB
  __shared__ __align__(16) _Float16 sWT1[64 * 128];   // 16 KB
  __shared__ __align__(16) _Float16 sWT2[32 * 64];    //  4 KB
  __shared__ __align__(16) _Float16 sWT3[16 * 32];    //  1 KB
  __shared__ float sB1[64], sB2[32], sB3[16];
  __shared__ float sTW0[16 * 8], sTB0[8];
  __shared__ float sTW1[8 * 4],  sTB1[4];
  __shared__ float sTW2[4 * NUM_REL], sTB2[NUM_REL];
  // ---- per-wave fused workspace (8 KB each):
  //   [0    .. 4095] actA : 16x128 f16   (layer0 out / layer2 out as 16x32)
  //   [4096 .. 6143] actB : 16x64  f16   (layer1 out)
  //   [6144 .. 8191] stage: 32x16  f32   (layer3 out)
  //   [0    .. 6527] ostage: 32x51 f32   (tail out; overlays the above, see ordering)
  __shared__ __align__(16) unsigned char sWS[NWAVES][8192];

  const int tid = threadIdx.x;

  // ---- cooperative weight staging: f32 -> f16, transposed [N][K] ----
  for (int i = tid; i < 128 * 32; i += NTHREADS) {
    int n = i >> 5, k = i & 31;
    float w = (k < 16) ? oW0[k * 128 + n] : (k == 16 ? ob0[n] : 0.f);
    sWT0[i] = (_Float16)w;
  }
  for (int i = tid; i < 64 * 128; i += NTHREADS) {
    int n = i >> 7, k = i & 127;
    sWT1[i] = (_Float16)oW1[k * 64 + n];
  }
  for (int i = tid; i < 32 * 64; i += NTHREADS) {
    int n = i >> 6, k = i & 63;
    sWT2[i] = (_Float16)oW2[k * 32 + n];
  }
  for (int i = tid; i < 16 * 32; i += NTHREADS) {
    int n = i >> 5, k = i & 31;
    sWT3[i] = (_Float16)oW3[k * 16 + n];
  }
  for (int i = tid; i < 64;  i += NTHREADS) sB1[i] = ob1[i];
  for (int i = tid; i < 32;  i += NTHREADS) sB2[i] = ob2[i];
  for (int i = tid; i < 16;  i += NTHREADS) sB3[i] = ob3[i];
  for (int i = tid; i < 16 * 8; i += NTHREADS) sTW0[i] = rW0[i];
  for (int i = tid; i < 8;      i += NTHREADS) sTB0[i] = rb0[i];
  for (int i = tid; i < 8 * 4;  i += NTHREADS) sTW1[i] = rW1[i];
  for (int i = tid; i < 4;      i += NTHREADS) sTB1[i] = rb1[i];
  for (int i = tid; i < 4 * NUM_REL; i += NTHREADS) sTW2[i] = rW2[i];
  for (int i = tid; i < NUM_REL;     i += NTHREADS) sTB2[i] = rb2[i];
  __syncthreads();

  const int lane = tid & 31;
  const int wv   = tid >> 5;
  const int mrow = lane & 15;   // M (or N) index within 16
  const int kh   = lane >> 4;   // K-half selector of the 16-bit fragment layout
  _Float16* actA   = (_Float16*)(sWS[wv]);
  _Float16* actB   = (_Float16*)(sWS[wv] + 4096);
  float*    stage  = (float*)   (sWS[wv] + 6144);
  float*    ostage = (float*)   (sWS[wv]);

  const int waveId     = blockIdx.x * NWAVES + wv;
  const int waveStride = gridDim.x * NWAVES;

  for (int tile = waveId; tile < N_TILES; tile += waveStride) {
    const int base = tile * 32;

#pragma unroll
    for (int mt = 0; mt < 2; ++mt) {
      // ---- gather + build layer-0 A fragment directly from global ----
      // lane m (0..15): src object feats 0..7 ; lane m+16: dst feats 8..15.
      // K=16 carries 1.0 (kh==0 lanes) to drive the bias row of WT0.
      const int edge = base + mt * 16 + mrow;
      const int oidx = pairs[2 * edge + kh];
      const float4* op = (const float4*)(obj + (size_t)oidx * 8);
      const float4 f0 = op[0], f1 = op[1];
      V16U a0u;
      a0u.v[0] = (_Float16)f0.x; a0u.v[1] = (_Float16)f0.y;
      a0u.v[2] = (_Float16)f0.z; a0u.v[3] = (_Float16)f0.w;
      a0u.v[4] = (_Float16)f1.x; a0u.v[5] = (_Float16)f1.y;
      a0u.v[6] = (_Float16)f1.z; a0u.v[7] = (_Float16)f1.w;
      a0u.v[8] = (kh == 0) ? (_Float16)1.f : (_Float16)0.f;   // K=16 / K=24
#pragma unroll
      for (int e = 9; e < 16; ++e) a0u.v[e] = (_Float16)0.f;
      const v16h a0 = a0u.v;

      // ---- layer 0: [16x32] x [32x128] (bias folded), ReLU -> actA[16][128] ----
#pragma unroll
      for (int nb = 0; nb < 8; ++nb) {
        v16h b = load_frag(sWT0 + (nb * 16 + mrow) * 32, kh * 8);
        v8f  c = splat8(0.f);
        c = __builtin_amdgcn_wmma_f32_16x16x32_f16(false, a0, false, b,
                                                   (short)0, c, false, false);
#pragma unroll
        for (int r = 0; r < 8; ++r) {
          float f = c[r] > 0.f ? c[r] : 0.f;
          actA[(r + 8 * kh) * 128 + nb * 16 + mrow] = (_Float16)f;
        }
      }
      lds_fence();

      // ---- layer 1: 128 -> 64, ReLU -> actB[16][64] ----
#pragma unroll
      for (int nb = 0; nb < 4; ++nb) {
        v8f c = splat8(sB1[nb * 16 + mrow]);
#pragma unroll
        for (int kb = 0; kb < 4; ++kb) {
          v16h a = load_frag(actA + mrow * 128 + kb * 32, kh * 8);
          v16h b = load_frag(sWT1 + (nb * 16 + mrow) * 128 + kb * 32, kh * 8);
          c = __builtin_amdgcn_wmma_f32_16x16x32_f16(false, a, false, b,
                                                     (short)0, c, false, false);
        }
#pragma unroll
        for (int r = 0; r < 8; ++r) {
          float f = c[r] > 0.f ? c[r] : 0.f;
          actB[(r + 8 * kh) * 64 + nb * 16 + mrow] = (_Float16)f;
        }
      }
      lds_fence();

      // ---- layer 2: 64 -> 32, ReLU -> actA reused as [16][32] ----
#pragma unroll
      for (int nb = 0; nb < 2; ++nb) {
        v8f c = splat8(sB2[nb * 16 + mrow]);
#pragma unroll
        for (int kb = 0; kb < 2; ++kb) {
          v16h a = load_frag(actB + mrow * 64 + kb * 32, kh * 8);
          v16h b = load_frag(sWT2 + (nb * 16 + mrow) * 64 + kb * 32, kh * 8);
          c = __builtin_amdgcn_wmma_f32_16x16x32_f16(false, a, false, b,
                                                     (short)0, c, false, false);
        }
#pragma unroll
        for (int r = 0; r < 8; ++r) {
          float f = c[r] > 0.f ? c[r] : 0.f;
          actA[(r + 8 * kh) * 32 + nb * 16 + mrow] = (_Float16)f;
        }
      }
      lds_fence();

      // ---- layer 3: 32 -> 16 (no ReLU) -> f32 stage[32][16] ----
      {
        v16h a = load_frag(actA + mrow * 32, kh * 8);
        v16h b = load_frag(sWT3 + mrow * 32, kh * 8);
        v8f  c = splat8(sB3[mrow]);
        c = __builtin_amdgcn_wmma_f32_16x16x32_f16(false, a, false, b,
                                                   (short)0, c, false, false);
#pragma unroll
        for (int r = 0; r < 8; ++r)
          stage[(mt * 16 + r + 8 * kh) * 16 + mrow] = c[r];
      }
      lds_fence();
    } // mt

    // ---- scalar tail MLP (16->8->4->51) — one edge per lane ----
    {
      float h[16];
#pragma unroll
      for (int k = 0; k < 16; ++k) h[k] = stage[lane * 16 + k];
      lds_fence();   // h in registers before ostage overlays stage head
      float t0[8];
#pragma unroll
      for (int j = 0; j < 8; ++j) {
        float s = sTB0[j];
#pragma unroll
        for (int k = 0; k < 16; ++k) s += h[k] * sTW0[k * 8 + j];
        t0[j] = s > 0.f ? s : 0.f;
      }
      float t1[4];
#pragma unroll
      for (int j = 0; j < 4; ++j) {
        float s = sTB1[j];
#pragma unroll
        for (int k = 0; k < 8; ++k) s += t0[k] * sTW1[k * 4 + j];
        t1[j] = s > 0.f ? s : 0.f;
      }
#pragma unroll
      for (int j = 0; j < NUM_REL; ++j) {
        float s = sTB2[j];
#pragma unroll
        for (int k = 0; k < 4; ++k) s += t1[k] * sTW2[k * NUM_REL + j];
        ostage[lane * NUM_REL + j] = s;
      }
      lds_fence();

      // ---- coalesced tile store: 32*51 f32 = 408 float4, 16B-aligned ----
      const float4* os4 = (const float4*)ostage;
      float4* og4 = (float4*)(out + (size_t)base * NUM_REL);
      for (int i = lane; i < (32 * NUM_REL) / 4; i += 32)
        og4[i] = os4[i];
      lds_fence();   // ostage reads done before next tile's actA writes
    }
  } // tile loop
}

// Pass-through of the int32 `relations` tensor into the tail of d_out (vectorized).
__global__ void copy_rel_kernel(const int4* __restrict__ rel,
                                int4* __restrict__ dst, int n4) {
  int i = blockIdx.x * blockDim.x + threadIdx.x;
  const int stride = gridDim.x * blockDim.x;
  for (; i < n4; i += stride) dst[i] = rel[i];
}

extern "C" void kernel_launch(void* const* d_in, const int* in_sizes, int n_in,
                              void* d_out, int out_size, void* d_ws, size_t ws_size,
                              hipStream_t stream) {
  const float* obj   = (const float*)d_in[0];
  const int*   rel   = (const int*)  d_in[1];
  const int*   pairs = (const int*)  d_in[2];
  const float* oW0 = (const float*)d_in[3];  const float* ob0 = (const float*)d_in[4];
  const float* oW1 = (const float*)d_in[5];  const float* ob1 = (const float*)d_in[6];
  const float* oW2 = (const float*)d_in[7];  const float* ob2 = (const float*)d_in[8];
  const float* oW3 = (const float*)d_in[9];  const float* ob3 = (const float*)d_in[10];
  const float* rW0 = (const float*)d_in[11]; const float* rb0 = (const float*)d_in[12];
  const float* rW1 = (const float*)d_in[13]; const float* rb1 = (const float*)d_in[14];
  const float* rW2 = (const float*)d_in[15]; const float* rb2 = (const float*)d_in[16];

  float* outF = (float*)d_out;
  int*   relO = (int*)d_out + (long long)E_EDGES * NUM_REL;

  sg_mlp_kernel<<<2048, NTHREADS, 0, stream>>>(
      obj, pairs, oW0, ob0, oW1, ob1, oW2, ob2, oW3, ob3,
      rW0, rb0, rW1, rb1, rW2, rb2, outF);
  copy_rel_kernel<<<2048, 256, 0, stream>>>((const int4*)rel, (int4*)relO,
                                            E_EDGES / 4);
}
